// ResidualVectorQuantizer_27908697489533
// MI455X (gfx1250) — matrix operand order repaired and checked by reading, not compiled, not run on hardware
//
#include <hip/hip_runtime.h>
#include <stdint.h>

// Problem constants (from reference)
#define NQ      8
#define KCB     1024
#define DIM     256
#define BT      32768      // 32*1024 tokens
#define M_TILE  128        // tokens per workgroup in argmin kernel
#define CHUNK   64         // codebook rows staged in LDS per iteration

typedef __attribute__((ext_vector_type(16))) __bf16    v16bf;
typedef __attribute__((ext_vector_type(8)))  float     v8f;
typedef __attribute__((ext_vector_type(8)))  uint32_t  v8u;

// ---- fp32 -> bf16 (RNE) helpers, header-independent ----
__device__ __forceinline__ uint32_t f2bf(float x) {
  union { float f; uint32_t u; } c; c.f = x;
  uint32_t u = c.u;
  u += 0x7FFFu + ((u >> 16) & 1u);
  return u >> 16;
}
__device__ __forceinline__ float bfhi2f(uint32_t h) {
  union { uint32_t u; float f; } c; c.u = h << 16;
  return c.f;
}
// split two fp32 into packed bf16 hi-plane word and lo-plane (residual) word
__device__ __forceinline__ void split2(float a, float b, uint32_t& hi, uint32_t& lo) {
  uint32_t ha = f2bf(a), hb = f2bf(b);
  float la = a - bfhi2f(ha);
  float lb = b - bfhi2f(hb);
  hi = ha | (hb << 16);
  lo = f2bf(la) | (f2bf(lb) << 16);
}

// ---------------------------------------------------------------------------
// ||c||^2 per codebook row: 8 waves/block, one wave per code row.
// ---------------------------------------------------------------------------
__global__ void rvq_c2(const float* __restrict__ cbs, float* __restrict__ c2g) {
  const int wave = threadIdx.x >> 5;
  const int lane = threadIdx.x & 31;
  const int code = blockIdx.x * 8 + wave;      // 0 .. NQ*KCB-1
  const float* row = cbs + (size_t)code * DIM;
  float s = 0.f;
  #pragma unroll
  for (int i = 0; i < 8; ++i) {
    float v = row[lane * 8 + i];
    s += v * v;
  }
  #pragma unroll
  for (int off = 16; off >= 1; off >>= 1) s += __shfl_xor(s, off, 32);
  if (lane == 0) c2g[code] = s;
}

// ---------------------------------------------------------------------------
// Argmin kernel: one quantizer step's distance GEMM + row argmin.
// Split-bf16 (hi+lo) WMMA with three independent accumulator chains:
//   acc = Ah*Bh + Ah*Bl + Al*Bh  (~fp32 accuracy, 3-way XDL pipelining)
// dist(t,k) = ||c_k||^2 - 2 * r_t.c_k   (||r||^2 dropped: constant per row)
// residual r = z - quantized_so_far (recomputed, no residual buffer).
// ---------------------------------------------------------------------------
__global__ __launch_bounds__(256, 1)
void rvq_argmin(const float* __restrict__ z, const float* __restrict__ quant,
                const float* __restrict__ cb, const float* __restrict__ c2,
                float* __restrict__ idx_out, int first) {
  extern __shared__ char smem[];                       // 64KB dynamic LDS
  uint32_t* bhi = (uint32_t*)smem;                     // CHUNK x DIM bf16 hi
  uint32_t* blo = (uint32_t*)(smem + CHUNK * DIM * 2); // CHUNK x DIM bf16 lo

  const int tid  = threadIdx.x;
  const int wave = tid >> 5;
  const int lane = tid & 31;
  const int half = lane >> 4;
  const int l16  = lane & 15;
  const int m0   = blockIdx.x * M_TILE;
  const int arow = m0 + wave * 16 + l16;   // A-operand row for this lane (M = lane%16)

  // ---- Hoist A fragments (hi/lo planes, all K=0..255) into registers ----
  // ISA 16-bit A layout: lane<16: VGPR v -> K = 2v (+8 if v>=4); lanes>=16: +8.
  v8u a_hi[8], a_lo[8];
  #pragma unroll
  for (int ks = 0; ks < 8; ++ks) {
    #pragma unroll
    for (int v = 0; v < 8; ++v) {
      const int K = ks * 32 + 2 * v + ((v >= 4) ? 8 : 0) + 8 * half;
      const float2 zz = *(const float2*)(z + (size_t)arow * DIM + K);
      float r0 = zz.x, r1 = zz.y;
      if (!first) {
        const float2 qq = *(const float2*)(quant + (size_t)arow * DIM + K);
        r0 -= qq.x; r1 -= qq.y;
      }
      uint32_t h, l;
      split2(r0, r1, h, l);
      a_hi[ks][v] = h;
      a_lo[ks][v] = l;
    }
  }

  float minv[8];
  int   mini[8];
  #pragma unroll
  for (int j = 0; j < 8; ++j) { minv[j] = 3.4e38f; mini[j] = 0; }

  for (int c0 = 0; c0 < KCB; c0 += CHUNK) {
    __syncthreads();
    // Stage CHUNK codebook rows (fp32 -> bf16 hi/lo) into LDS, coalesced.
    #pragma unroll
    for (int it = 0; it < (CHUNK * DIM / 4 / 256); ++it) {
      const int f = tid + it * 256;        // float4 index within chunk
      const int e = f * 4;                 // element index
      const float4 cc = *(const float4*)(cb + (size_t)c0 * DIM + e);
      uint32_t h0, l0, h1, l1;
      split2(cc.x, cc.y, h0, l0);
      split2(cc.z, cc.w, h1, l1);
      bhi[(e >> 1) + 0] = h0;  bhi[(e >> 1) + 1] = h1;
      blo[(e >> 1) + 0] = l0;  blo[(e >> 1) + 1] = l1;
    }
    __syncthreads();

    for (int nb = 0; nb < CHUNK; nb += 16) {
      // Three independent accumulator chains for XDL pipelining.
      v8f acc_hh = {}, acc_hl = {}, acc_lh = {};
      // B fragment is two 16-byte runs per lane in LDS, 32 bytes apart:
      // u32 base index = (col)*128 + ks*16 + 4*half   (K = 2v + 8*half, +8 if v>=4)
      const int base = (nb + l16) * (DIM / 2) + 4 * half;
      #pragma unroll
      for (int ks = 0; ks < 8; ++ks) {
        const int o = base + ks * 16;
        const uint4 h0 = *(const uint4*)(bhi + o), h1 = *(const uint4*)(bhi + o + 8);
        const uint4 q0 = *(const uint4*)(blo + o), q1 = *(const uint4*)(blo + o + 8);
        const v8u uh = {h0.x, h0.y, h0.z, h0.w, h1.x, h1.y, h1.z, h1.w};
        const v8u ul = {q0.x, q0.y, q0.z, q0.w, q1.x, q1.y, q1.z, q1.w};
        v16bf bh = __builtin_bit_cast(v16bf, uh);
        v16bf bl = __builtin_bit_cast(v16bf, ul);
        v16bf ah = __builtin_bit_cast(v16bf, a_hi[ks]);
        v16bf al = __builtin_bit_cast(v16bf, a_lo[ks]);
        acc_hh = __builtin_amdgcn_wmma_f32_16x16x32_bf16(false, ah, false, bh, (short)0, acc_hh, false, false);
        acc_hl = __builtin_amdgcn_wmma_f32_16x16x32_bf16(false, ah, false, bl, (short)0, acc_hl, false, false);
        acc_lh = __builtin_amdgcn_wmma_f32_16x16x32_bf16(false, al, false, bh, (short)0, acc_lh, false, false);
      }
      // C layout: VGPR j -> (row = j + 8*half, col = lane%16)
      const int   col = c0 + nb + l16;
      const float c2v = c2[col];
      #pragma unroll
      for (int j = 0; j < 8; ++j) {
        const float d = c2v - 2.0f * (acc_hh[j] + acc_hl[j] + acc_lh[j]);
        if (d < minv[j]) { minv[j] = d; mini[j] = col; }
      }
    }
  }

  // Reduce argmin across the 16 lanes (columns) within each half-wave.
  #pragma unroll
  for (int j = 0; j < 8; ++j) {
    #pragma unroll
    for (int off = 8; off >= 1; off >>= 1) {
      const float ov = __shfl_xor(minv[j], off, 16);
      const int   oi = __shfl_xor(mini[j], off, 16);
      if (ov < minv[j]) { minv[j] = ov; mini[j] = oi; }
    }
  }
  if (l16 == 0) {
    #pragma unroll
    for (int j = 0; j < 8; ++j) {
      const int r = m0 + wave * 16 + j + 8 * half;   // C-row -> token id
      idx_out[r] = (float)mini[j];
    }
  }
}

// ---------------------------------------------------------------------------
// Update kernel: gather cb[idx], quantized += q, accumulate 1.25*mean((q-r)^2).
// One wave per token; lane covers 8 dims (2x float4).
// ---------------------------------------------------------------------------
__global__ __launch_bounds__(256, 1)
void rvq_update(const float* __restrict__ z, float* __restrict__ quant,
                const float* __restrict__ cb, const float* __restrict__ idx_in,
                float* __restrict__ loss_acc, int first) {
  const int wave = threadIdx.x >> 5;
  const int lane = threadIdx.x & 31;
  const int t    = blockIdx.x * 8 + wave;
  const int k    = (int)idx_in[t];
  const float* crow = cb + (size_t)k * DIM;

  float sumsq = 0.f;
  #pragma unroll
  for (int i = 0; i < 2; ++i) {
    const int d = lane * 8 + i * 4;
    const size_t o = (size_t)t * DIM + d;
    const float4 zz = *(const float4*)(z + o);
    float4 qv;
    if (first) { qv = make_float4(0.f, 0.f, 0.f, 0.f); }
    else       { qv = *(const float4*)(quant + o); }
    const float4 cc = *(const float4*)(crow + d);
    const float d0 = cc.x - (zz.x - qv.x);
    const float d1 = cc.y - (zz.y - qv.y);
    const float d2 = cc.z - (zz.z - qv.z);
    const float d3 = cc.w - (zz.w - qv.w);
    sumsq += d0 * d0 + d1 * d1 + d2 * d2 + d3 * d3;
    const float4 nq = make_float4(qv.x + cc.x, qv.y + cc.y, qv.z + cc.z, qv.w + cc.w);
    *(float4*)(quant + o) = nq;
  }
  #pragma unroll
  for (int off = 16; off >= 1; off >>= 1) sumsq += __shfl_xor(sumsq, off, 32);
  if (lane == 0) atomicAdd(loss_acc, sumsq);
}

__global__ void rvq_init(float* __restrict__ loss_acc) {
  if (threadIdx.x == 0) *loss_acc = 0.f;
}

__global__ void rvq_finalize(const float* __restrict__ loss_acc, float* __restrict__ out) {
  if (threadIdx.x == 0) {
    // sum_q (codebook_loss + 0.25*commit) == 1.25 * sum_sq / (B*T*D)
    *out = 1.25f * (*loss_acc) / (float)(BT * DIM);
  }
}

// ---------------------------------------------------------------------------
extern "C" void kernel_launch(void* const* d_in, const int* in_sizes, int n_in,
                              void* d_out, int out_size, void* d_ws, size_t ws_size,
                              hipStream_t stream) {
  (void)in_sizes; (void)n_in; (void)out_size; (void)ws_size;
  const float* z   = (const float*)d_in[0];              // [32,1024,256]
  const float* cbs = (const float*)d_in[1];              // [8,1024,256]

  float* quant   = (float*)d_out;                        // [32,1024,256]
  float* idxs    = quant + (size_t)BT * DIM;             // [8,32,1024] as float
  float* lossout = idxs + (size_t)NQ * BT;               // scalar

  float* c2g     = (float*)d_ws;                         // [8,1024] ||c||^2
  float* lossacc = c2g + NQ * KCB;                       // scalar accumulator

  rvq_init<<<1, 1, 0, stream>>>(lossacc);
  rvq_c2<<<NQ * KCB / 8, 256, 0, stream>>>(cbs, c2g);

  for (int q = 0; q < NQ; ++q) {
    const float* cbq = cbs + (size_t)q * KCB * DIM;
    rvq_argmin<<<BT / M_TILE, 256, CHUNK * DIM * 4, stream>>>(
        z, quant, cbq, c2g + q * KCB, idxs + (size_t)q * BT, q == 0);
    rvq_update<<<BT / 8, 256, 0, stream>>>(
        z, quant, cbq, idxs + (size_t)q * BT, lossacc, q == 0);
  }
  rvq_finalize<<<1, 1, 0, stream>>>(lossacc, lossout);
}
